// MambaBlock_70514773066075
// MI455X (gfx1250) — matrix operand order repaired and checked by reading, compiled, not run on hardware
//
#include <hip/hip_runtime.h>
#include <hip/hip_bf16.h>
#include <stdint.h>

// ---------------- problem constants ----------------
#define BB   2
#define TT   8
#define HH   32
#define WW   32
#define CC   192
#define DIM  384           // d_inner
#define NST  16            // d_state
#define KDIR 4
#define DTRK 12
#define LTOK 77
#define HID  768
#define LL   (TT*HH*WW)    // 8192
#define BL   (BB*LL)       // 16384
#define C2   (2*DIM)       // 768
#define XPN  48            // DTR + 2*N = 44, padded to 48 (x_proj output stride)
#define WPAD 64            // weight rows padded to 64 for the 16x64 wave tile
#define SCANTB 128         // scan block size (DIM split across DIM/SCANTB blocks)

typedef __attribute__((ext_vector_type(16))) __bf16 v16bf;
typedef __attribute__((ext_vector_type(8)))  float  v8f;

union FragBF { v16bf v; unsigned u[8]; };

__device__ __forceinline__ unsigned short f2bf(float f) {
  unsigned u = __float_as_uint(f);
  unsigned r = u + 0x7FFFu + ((u >> 16) & 1u);   // round-to-nearest-even
  return (unsigned short)(r >> 16);
}
__device__ __forceinline__ float sigm(float x)  { return 1.f / (1.f + __expf(-x)); }
__device__ __forceinline__ float siluf(float x) { return x * sigm(x); }
__device__ __forceinline__ float geluf(float x) {
  float x3 = x * x * x;
  return 0.5f * x * (1.f + tanhf(0.7978845608028654f * (x + 0.044715f * x3)));
}
__device__ __forceinline__ float softplusf(float x) {
  return (x > 20.f) ? x : log1pf(__expf(x));
}
// scan index l of direction k -> canonical (t,h,w) flat index.
// Same mapping serves the u-gather, the x_proj A-gather, and the merge scatter.
__device__ __forceinline__ int scan_src_index(int k, int l) {
  int j = (k >= 2) ? (LL - 1 - l) : l;
  if ((k & 1) == 0) return j;
  int t = j / (WW * HH);
  int r = j - t * (WW * HH);
  int w = r / HH;
  int h = r - w * HH;
  return (t * HH + h) * WW + w;
}

__device__ __forceinline__ void load_frag_a(FragBF& f, const unsigned short* p) {
  // A fragment (16x32 bf16): per-lane K pairs 0-7/16-23 (half 0) or 8-15/24-31 (half 1)
  uint4 a0 = *reinterpret_cast<const uint4*>(p);
  uint4 a1 = *reinterpret_cast<const uint4*>(p + 16);
  f.u[0] = a0.x; f.u[1] = a0.y; f.u[2] = a0.z; f.u[3] = a0.w;
  f.u[4] = a1.x; f.u[5] = a1.y; f.u[6] = a1.z; f.u[7] = a1.w;
}
__device__ __forceinline__ void load_frag_b(FragBF& f, const unsigned short* p) {
  // B fragment (32x16 bf16): per-lane K 0-7/16-23 (half 0) or 8-15/24-31 (half 1)
  uint4 b0 = *reinterpret_cast<const uint4*>(p);
  uint4 b1 = *reinterpret_cast<const uint4*>(p + 16);
  f.u[0] = b0.x; f.u[1] = b0.y; f.u[2] = b0.z; f.u[3] = b0.w;
  f.u[4] = b1.x; f.u[5] = b1.y; f.u[6] = b1.z; f.u[7] = b1.w;
}

// ---------------- bf16 WMMA GEMM, 16x64 tile per wave (4 accumulators) ----------------
// Y[M,N] = act(A[M,K] @ Wt[N,K]^T + bias) (+ resid).
// Wt must be valid (zero-padded) up to ceil(N/64)*64 rows.
__global__ void gemm_bf16_wmma(const unsigned short* __restrict__ A,
                               const unsigned short* __restrict__ Wt,
                               const float* __restrict__ bias,
                               const float* __restrict__ resid,
                               float* __restrict__ Y,
                               int M, int N, int K, int act)
{
  const int lane = threadIdx.x & 31;
  const int wave = threadIdx.x >> 5;
  const int tg   = ((N >> 4) + 3) >> 2;        // 64-wide column groups
  const int tile = blockIdx.x * (blockDim.x >> 5) + wave;
  const int total = (M >> 4) * tg;
  if (tile >= total) return;                    // wave-uniform: EXEC all-ones for WMMA
  const int m0  = (tile / tg) << 4;
  const int ng0 = (tile % tg) << 6;             // first column of this 64-wide group
  const int r    = lane & 15;
  const int half = lane >> 4;

  const unsigned short* pa = A  + (size_t)(m0 + r) * K + half * 8;
  const unsigned short* pb = Wt + (size_t)(ng0 + r) * K + half * 8;
  const size_t bstep = (size_t)16 * K;          // next 16-column tile of B

  v8f acc0 = {0.f,0.f,0.f,0.f,0.f,0.f,0.f,0.f};
  v8f acc1 = acc0, acc2 = acc0, acc3 = acc0;
  for (int k0 = 0; k0 < K; k0 += 32) {
    __builtin_prefetch(pa + k0 + 256, 0, 3);    // global_prefetch_b8
    __builtin_prefetch(pb + k0 + 256, 0, 3);
    FragBF fa, fb0, fb1, fb2, fb3;
    load_frag_a(fa, pa + k0);
    load_frag_b(fb0, pb + k0);
    load_frag_b(fb1, pb + bstep + k0);
    load_frag_b(fb2, pb + 2 * bstep + k0);
    load_frag_b(fb3, pb + 3 * bstep + k0);
    acc0 = __builtin_amdgcn_wmma_f32_16x16x32_bf16(false, fa.v, false, fb0.v, (short)0, acc0, false, false);
    acc1 = __builtin_amdgcn_wmma_f32_16x16x32_bf16(false, fa.v, false, fb1.v, (short)0, acc1, false, false);
    acc2 = __builtin_amdgcn_wmma_f32_16x16x32_bf16(false, fa.v, false, fb2.v, (short)0, acc2, false, false);
    acc3 = __builtin_amdgcn_wmma_f32_16x16x32_bf16(false, fa.v, false, fb3.v, (short)0, acc3, false, false);
  }
  // epilogue: 4 column tiles, predicated on col < N (stores only; WMMA was unpredicated)
  v8f accs[4] = {acc0, acc1, acc2, acc3};
#pragma unroll
  for (int j = 0; j < 4; ++j) {
    const int col = ng0 + j * 16 + r;
    if (col >= N) continue;
    const float bv = bias ? bias[col] : 0.f;
#pragma unroll
    for (int i = 0; i < 8; ++i) {
      const int row = m0 + i + half * 8;        // C/D layout: VGPR i -> M=i / M=8+i
      float v = accs[j][i] + bv;
      if (act == 1) v = geluf(v);
      if (resid) v += resid[(size_t)row * N + col];
      Y[(size_t)row * N + col] = v;
    }
  }
}

// ---------------- depthwise 3x3x3 conv + residual (CPE), channels-last ----------------
__global__ void cpe_kernel(const float* __restrict__ X,
                           const float* __restrict__ Wc,
                           const float* __restrict__ Bc,
                           float* __restrict__ O, int total)
{
  int idx = blockIdx.x * blockDim.x + threadIdx.x;
  if (idx >= total) return;
  int c = idx % CC;
  int l = (idx / CC) % LL;
  int b = idx / (CC * LL);
  int t = l / (HH * WW);
  int r = l - t * (HH * WW);
  int h = r / WW;
  int w = r - h * WW;
  float acc = Bc[c];
  for (int dt = -1; dt <= 1; ++dt) {
    int t2 = t + dt; if (t2 < 0 || t2 >= TT) continue;
    for (int dh = -1; dh <= 1; ++dh) {
      int h2 = h + dh; if (h2 < 0 || h2 >= HH) continue;
      for (int dw = -1; dw <= 1; ++dw) {
        int w2 = w + dw; if (w2 < 0 || w2 >= WW) continue;
        int l2  = (t2 * HH + h2) * WW + w2;
        int tap = ((dt + 1) * 3 + (dh + 1)) * 3 + (dw + 1);
        acc += X[((size_t)b * LL + l2) * CC + c] * Wc[c * 27 + tap];
      }
    }
  }
  O[idx] = X[idx] + acc;
}

// ---------------- depthwise conv on xi (= xz[:,:DIM] + cond) + SiLU -> XC[B,DIM,L] ----
__global__ void conv_silu_kernel(const float* __restrict__ XZ,
                                 const float* __restrict__ COND,
                                 const float* __restrict__ Wc,
                                 const float* __restrict__ Bc,
                                 float* __restrict__ XC, int total)
{
  int idx = blockIdx.x * blockDim.x + threadIdx.x;
  if (idx >= total) return;
  int l = idx % LL;
  int d = (idx / LL) % DIM;
  int b = idx / (LL * DIM);
  int t = l / (HH * WW);
  int r = l - t * (HH * WW);
  int h = r / WW;
  int w = r - h * WW;
  float cnd = COND[b * DIM + d];
  float acc = Bc[d];
  for (int dt = -1; dt <= 1; ++dt) {
    int t2 = t + dt; if (t2 < 0 || t2 >= TT) continue;
    for (int dh = -1; dh <= 1; ++dh) {
      int h2 = h + dh; if (h2 < 0 || h2 >= HH) continue;
      for (int dw = -1; dw <= 1; ++dw) {
        int w2 = w + dw; if (w2 < 0 || w2 >= WW) continue;
        int l2  = (t2 * HH + h2) * WW + w2;
        int tap = ((dt + 1) * 3 + (dh + 1)) * 3 + (dw + 1);
        acc += (XZ[((size_t)b * LL + l2) * C2 + d] + cnd) * Wc[d * 27 + tap];
      }
    }
  }
  XC[idx] = siluf(acc);
}

// ---------------- LayerNorm (wave per row) -> bf16 output ----------------
__global__ void ln_bf16_kernel(const float* __restrict__ X,
                               const float* __restrict__ G,
                               const float* __restrict__ Bt,
                               unsigned short* __restrict__ O,
                               int rows, int CN)
{
  int wid  = (blockIdx.x * blockDim.x + threadIdx.x) >> 5;
  int lane = threadIdx.x & 31;
  if (wid >= rows) return;
  const float* xr = X + (size_t)wid * CN;
  const int cnt = CN >> 5;
  float vals[12];
  float s = 0.f, sq = 0.f;
  for (int i = 0; i < cnt; ++i) { float v = xr[lane + (i << 5)]; vals[i] = v; s += v; sq += v * v; }
  for (int o = 16; o > 0; o >>= 1) { s += __shfl_xor(s, o, 32); sq += __shfl_xor(sq, o, 32); }
  float m  = s / (float)CN;
  float rs = rsqrtf(sq / (float)CN - m * m + 1e-6f);
  for (int i = 0; i < cnt; ++i) {
    int c = lane + (i << 5);
    O[(size_t)wid * CN + c] = f2bf((vals[i] - m) * rs * G[c] + Bt[c]);
  }
}

// ---------------- out_norm over DIM (strided) + SiLU(z) gate -> bf16 ----------------
__global__ void outnorm_gate_kernel(const float* __restrict__ YM,
                                    const float* __restrict__ XZ,
                                    const float* __restrict__ G,
                                    const float* __restrict__ Bt,
                                    unsigned short* __restrict__ O, int rows)
{
  int wid  = (blockIdx.x * blockDim.x + threadIdx.x) >> 5;
  int lane = threadIdx.x & 31;
  if (wid >= rows) return;
  int b  = wid / LL;
  int lc = wid - b * LL;
  const int cnt = DIM >> 5;   // 12
  float vals[12];
  float s = 0.f, sq = 0.f;
  for (int i = 0; i < cnt; ++i) {
    int d = lane + (i << 5);
    float v = YM[((size_t)b * DIM + d) * LL + lc];
    vals[i] = v; s += v; sq += v * v;
  }
  for (int o = 16; o > 0; o >>= 1) { s += __shfl_xor(s, o, 32); sq += __shfl_xor(sq, o, 32); }
  float m  = s / (float)DIM;
  float rs = rsqrtf(sq / (float)DIM - m * m + 1e-6f);
  for (int i = 0; i < cnt; ++i) {
    int d = lane + (i << 5);
    float yn = (vals[i] - m) * rs * G[d] + Bt[d];
    float z  = XZ[((size_t)b * LL + lc) * C2 + DIM + d];
    O[(size_t)wid * DIM + d] = f2bf(yn * siluf(z));
  }
}

// ---------------- gather permuted scan input -> bf16 A matrix for x_proj ----------------
__global__ void gather_bf16_kernel(const float* __restrict__ XC,
                                   unsigned short* __restrict__ O,
                                   int k, int total)
{
  int idx = blockIdx.x * blockDim.x + threadIdx.x;
  if (idx >= total) return;
  int d = idx % DIM;
  int l = (idx / DIM) % LL;
  int b = idx / (DIM * LL);
  int lc = scan_src_index(k, l);
  O[idx] = f2bf(XC[((size_t)b * DIM + d) * LL + lc]);
}

// ---------------- tiny text-conditioning kernels ----------------
__global__ void text_mean_kernel(const float* __restrict__ TXT, float* __restrict__ TM, int total) {
  int idx = blockIdx.x * blockDim.x + threadIdx.x;
  if (idx >= total) return;
  int c = idx % CC, b = idx / CC;
  float s = 0.f;
  for (int t = 0; t < LTOK; ++t) s += TXT[((size_t)b * LTOK + t) * CC + c];
  TM[idx] = s / (float)LTOK;
}
__global__ void cond_kernel(const float* __restrict__ TM, const float* __restrict__ Wp,
                            const float* __restrict__ Bp, float* __restrict__ CO, int total) {
  int idx = blockIdx.x * blockDim.x + threadIdx.x;
  if (idx >= total) return;
  int d = idx % DIM, b = idx / DIM;
  float s = Bp[d];
  for (int c = 0; c < CC; ++c) s += TM[b * CC + c] * Wp[d * CC + c];
  CO[idx] = siluf(s);
}

// ---------------- utility kernels ----------------
__global__ void zero_f32_kernel(float* P, int n)   { int i = blockIdx.x * blockDim.x + threadIdx.x; if (i < n) P[i] = 0.f; }
__global__ void zero_u16_kernel(unsigned short* P, int n) { int i = blockIdx.x * blockDim.x + threadIdx.x; if (i < n) P[i] = 0; }
__global__ void cvt_bf16_kernel(const float* X, unsigned short* O, int n) {
  int i = blockIdx.x * blockDim.x + threadIdx.x; if (i < n) O[i] = f2bf(X[i]);
}

// ---------------- selective scan ----------------
// grid = B*K*(DIM/SCANTB) blocks of SCANTB threads; one thread per (b,k,d) channel.
// Double-buffered LDS broadcast of the 44 shared per-step values -> 1 barrier/step.
__global__ void __launch_bounds__(SCANTB) scan_kernel(
    const float* __restrict__ XC,     // [B,DIM,L] conv output (u source, all 4 dirs)
    const float* __restrict__ XDBL,   // [K,B*L,48] x_proj output (dtr|B|C)
    const float* __restrict__ DTW,    // [K,DIM,DTR]
    const float* __restrict__ DTB,    // [K,DIM]
    const float* __restrict__ ALOG,   // [K,DIM,NST]
    const float* __restrict__ DS,     // [K,DIM]
    float* __restrict__ YM)           // [B,DIM,L] merged output (atomic accumulate)
{
  const int nq = DIM / SCANTB;
  const int dq = blockIdx.x % nq;
  const int bk = blockIdx.x / nq;
  const int b  = bk / KDIR;
  const int k  = bk % KDIR;
  const int d  = dq * SCANTB + threadIdx.x;
  __shared__ float sh[2][XPN];

  float aco[NST], hst[NST], dtw[DTRK];
#pragma unroll
  for (int n = 0; n < NST; ++n) {
    aco[n] = -__expf(ALOG[((size_t)k * DIM + d) * NST + n]);
    hst[n] = 0.f;
  }
#pragma unroll
  for (int r = 0; r < DTRK; ++r) dtw[r] = DTW[((size_t)k * DIM + d) * DTRK + r];
  const float dtb = DTB[k * DIM + d];
  const float dsv = DS[k * DIM + d];
  const float* xdk = XDBL + (size_t)k * BL * XPN + (size_t)b * LL * XPN;

  if (threadIdx.x < XPN) sh[0][threadIdx.x] = xdk[threadIdx.x];
  __syncthreads();

  for (int l = 0; l < LL; ++l) {
    const int cur = l & 1;
    if (l + 1 < LL && threadIdx.x < XPN)
      sh[cur ^ 1][threadIdx.x] = xdk[(size_t)(l + 1) * XPN + threadIdx.x];
    float dtr = dtb;
#pragma unroll
    for (int r = 0; r < DTRK; ++r) dtr += sh[cur][r] * dtw[r];
    const float dt = softplusf(dtr);
    const int lc = scan_src_index(k, l);
    const float u  = XC[((size_t)b * DIM + d) * LL + lc];
    const float du = dt * u;
    float y = 0.f;
#pragma unroll
    for (int n = 0; n < NST; ++n) {
      hst[n] = hst[n] * __expf(dt * aco[n]) + du * sh[cur][DTRK + n];
      y += hst[n] * sh[cur][DTRK + NST + n];
    }
    atomicAdd(&YM[((size_t)b * DIM + d) * LL + lc], y + dsv * u);
    __syncthreads();   // next buffer ready / current buffer free
  }
}

// ---------------- host orchestration ----------------
extern "C" void kernel_launch(void* const* d_in, const int* in_sizes, int n_in,
                              void* d_out, int out_size, void* d_ws, size_t ws_size,
                              hipStream_t stream)
{
  (void)in_sizes; (void)n_in; (void)out_size; (void)ws_size;
  const float* x          = (const float*)d_in[0];
  const float* text       = (const float*)d_in[1];
  const float* cpe1_w     = (const float*)d_in[2];
  const float* cpe1_b     = (const float*)d_in[3];
  const float* cpe2_w     = (const float*)d_in[4];
  const float* cpe2_b     = (const float*)d_in[5];
  const float* norm1_g    = (const float*)d_in[6];
  const float* norm1_b    = (const float*)d_in[7];
  const float* norm2_g    = (const float*)d_in[8];
  const float* norm2_b    = (const float*)d_in[9];
  const float* in_proj_w  = (const float*)d_in[10];
  const float* in_proj_b  = (const float*)d_in[11];
  const float* text_proj_w= (const float*)d_in[12];
  const float* text_proj_b= (const float*)d_in[13];
  const float* conv_w     = (const float*)d_in[14];
  const float* conv_b     = (const float*)d_in[15];
  const float* x_proj_w   = (const float*)d_in[16];
  const float* dt_proj_w  = (const float*)d_in[17];
  const float* dt_proj_b  = (const float*)d_in[18];
  const float* A_log      = (const float*)d_in[19];
  const float* Ds         = (const float*)d_in[20];
  const float* out_norm_g = (const float*)d_in[21];
  const float* out_norm_b = (const float*)d_in[22];
  const float* out_proj_w = (const float*)d_in[23];
  const float* fc1_w      = (const float*)d_in[24];
  const float* fc1_b      = (const float*)d_in[25];
  const float* fc2_w      = (const float*)d_in[26];
  const float* fc2_b      = (const float*)d_in[27];
  float* out = (float*)d_out;

  char* ws = (char*)d_ws;
  size_t off = 0;
  auto alloc = [&](size_t bytes) -> char* {
    char* p = ws + off;
    off += (bytes + 255) & ~(size_t)255;
    return p;
  };
  float* X1   = (float*)alloc((size_t)BL * CC * 4);          // x + cpe1 (later reused as X3)
  float* X2   = (float*)alloc((size_t)BL * CC * 4);          // x1 + ss2d
  float* XZ   = (float*)alloc((size_t)BL * C2 * 4);          // in_proj out; later fc1 out
  float* XC   = (float*)alloc((size_t)BB * DIM * LL * 4);    // conv+silu, [B,DIM,L]
  float* XDBL = (float*)alloc((size_t)KDIR * BL * XPN * 4);  // x_proj out, 4 dirs
  float* YM   = (float*)alloc((size_t)BB * DIM * LL * 4);    // merged scan out
  unsigned short* ABF = (unsigned short*)alloc((size_t)BL * C2 * 2); // bf16 GEMM A staging
  unsigned short* WBF = (unsigned short*)alloc((size_t)C2 * CC * 2); // bf16 weight staging
  float* TM   = (float*)alloc((size_t)BB * CC * 4);
  float* CO   = (float*)alloc((size_t)BB * DIM * 4);
  float* X3 = X1;  // X1 dead after out_proj residual; reuse for x2 + cpe2

  const int TB = 256;
  auto g1 = [&](int n) { return (n + TB - 1) / TB; };
  auto gemm = [&](const unsigned short* A, const unsigned short* Wt,
                  const float* bias, const float* resid, float* Y,
                  int M, int N, int K, int act) {
    int tg = ((N >> 4) + 3) >> 2;                 // 64-wide column groups
    int tiles = (M >> 4) * tg;
    gemm_bf16_wmma<<<(tiles + 7) / 8, 256, 0, stream>>>(A, Wt, bias, resid, Y, M, N, K, act);
  };
  auto cvt = [&](const float* src, unsigned short* dst, int n) {
    cvt_bf16_kernel<<<g1(n), TB, 0, stream>>>(src, dst, n);
  };

  // 1) x1 = x + cpe1(x)
  cpe_kernel<<<g1(BL * CC), TB, 0, stream>>>(x, cpe1_w, cpe1_b, X1, BL * CC);
  // 2-3) text conditioning
  text_mean_kernel<<<g1(BB * CC), TB, 0, stream>>>(text, TM, BB * CC);
  cond_kernel<<<g1(BB * DIM), TB, 0, stream>>>(TM, text_proj_w, text_proj_b, CO, BB * DIM);
  // 4-6) norm1 -> bf16, in_proj GEMM -> XZ
  ln_bf16_kernel<<<g1(BL * 32), TB, 0, stream>>>(X1, norm1_g, norm1_b, ABF, BL, CC);
  cvt(in_proj_w, WBF, C2 * CC);
  gemm(ABF, WBF, in_proj_b, nullptr, XZ, BL, C2, CC, 0);
  // 7) depthwise conv + cond + SiLU -> XC
  conv_silu_kernel<<<g1(BB * DIM * LL), TB, 0, stream>>>(XZ, CO, conv_w, conv_b, XC, BB * DIM * LL);
  // 8) per-direction x_proj GEMMs (weights zero-padded 44 -> 64 rows)
  for (int k = 0; k < KDIR; ++k) {
    gather_bf16_kernel<<<g1(BL * DIM), TB, 0, stream>>>(XC, ABF, k, BL * DIM);
    zero_u16_kernel<<<g1(WPAD * DIM), TB, 0, stream>>>(WBF, WPAD * DIM);
    cvt(x_proj_w + (size_t)k * 44 * DIM, WBF, 44 * DIM);
    gemm(ABF, WBF, nullptr, nullptr, XDBL + (size_t)k * BL * XPN, BL, XPN, DIM, 0);
  }
  // 9-10) selective scan with fused dt_proj/softplus and direction merge
  zero_f32_kernel<<<g1(BB * DIM * LL), TB, 0, stream>>>(YM, BB * DIM * LL);
  scan_kernel<<<BB * KDIR * (DIM / SCANTB), SCANTB, 0, stream>>>(
      XC, XDBL, dt_proj_w, dt_proj_b, A_log, Ds, YM);
  // 11-12) out_norm + SiLU(z) gate -> bf16, out_proj GEMM with fused residual(X1) -> X2
  outnorm_gate_kernel<<<g1(BL * 32), TB, 0, stream>>>(YM, XZ, out_norm_g, out_norm_b, ABF, BL);
  cvt(out_proj_w, WBF, CC * DIM);
  gemm(ABF, WBF, nullptr, X1, X2, BL, CC, DIM, 0);
  // 13) x3 = x2 + cpe2(x2)
  cpe_kernel<<<g1(BL * CC), TB, 0, stream>>>(X2, cpe2_w, cpe2_b, X3, BL * CC);
  // 14-17) MLP: norm2 -> fc1 (GELU) -> fc2 with fused residual(X3) -> out
  ln_bf16_kernel<<<g1(BL * 32), TB, 0, stream>>>(X3, norm2_g, norm2_b, ABF, BL, CC);
  cvt(fc1_w, WBF, HID * CC);
  gemm(ABF, WBF, fc1_b, nullptr, XZ, BL, HID, CC, 1);
  cvt(XZ, ABF, BL * HID);
  cvt(fc2_w, WBF, CC * HID);
  gemm(ABF, WBF, fc2_b, X3, out, BL, CC, HID, 0);
}
